// crosattn_spatial_q2k1v2_33174327394340
// MI455X (gfx1250) — matrix operand order repaired and checked by reading, not compiled, not run on hardware
//
#include <hip/hip_runtime.h>
#include <hip/hip_bf16.h>

// ---------------------------------------------------------------------------
// Cross-attention with softmax over the QUERY axis (axis=-2):
//   S[q,k] = scale * <q2[q,:], k1[k,:]>          (per b,h;  q2 == v2, from x2)
//   P[q,k] = exp(S[q,k]-m[k]) / l[k],  m/l reduced over q (column softmax)
//   out[q,:] = sum_k P[q,k] * v2[k,:]
// Two-pass scheme (recompute S; never materialize ctx in HBM; x1+x2 = 67 MB
// fit in the 192 MB L2, so cross-pass re-reads are L2-resident):
//   Kernel 1: column stats m[k], l[k]  -> d_ws (1 MB)
//   Kernel 2: recompute S, normalize, P @ V
// All matmuls: v_wmma_f32_16x16x32_bf16 (bf16 operands, f32 accumulate).
// ---------------------------------------------------------------------------

typedef __bf16 v16bf __attribute__((ext_vector_type(16)));
typedef __bf16 v2bf  __attribute__((ext_vector_type(2)));
typedef float  v8f   __attribute__((ext_vector_type(8)));
typedef float  v4f   __attribute__((ext_vector_type(4)));
typedef unsigned int v2u __attribute__((ext_vector_type(2)));
typedef unsigned int v4u __attribute__((ext_vector_type(4)));
typedef unsigned int v8u __attribute__((ext_vector_type(8)));

#define Bn 16
#define Hn 8
#define Nn 1024
#define Dd 64
#define Cc 512
#define LQ 72   // half-elem row stride of [token][d] LDS tiles (16B aligned rows)
#define LV 40   // half-elem row stride of [d][token] and P LDS tiles

__device__ __forceinline__ unsigned short f2bf(float f) {
  unsigned u = __builtin_bit_cast(unsigned, f);
  u += 0x7FFFu + ((u >> 16) & 1u);          // round-to-nearest-even
  return (unsigned short)(u >> 16);
}

// Pack two f32 -> packed bf16 pair (v_cvt_pk_bf16_f32 when available).
__device__ __forceinline__ unsigned pk2bf(float a, float b) {
#if __has_builtin(__builtin_amdgcn_cvt_pk_bf16_f32)
  v2bf p = __builtin_amdgcn_cvt_pk_bf16_f32(a, b);
  return __builtin_bit_cast(unsigned, p);
#else
  return (unsigned)f2bf(a) | ((unsigned)f2bf(b) << 16);
#endif
}

// Load a 16-half WMMA fragment as two b128 LDS reads (offsets in halves).
__device__ __forceinline__ v16bf ldsfrag(const unsigned short* p, int h0, int h1) {
  v4u lo = *(const v4u*)(p + h0);
  v4u hi = *(const v4u*)(p + h1);
  v8u w; w.lo = lo; w.hi = hi;
  return __builtin_bit_cast(v16bf, w);
}

__device__ __forceinline__ v8f wmma_bf16(v16bf a, v16bf b, v8f c) {
  return __builtin_amdgcn_wmma_f32_16x16x32_bf16(false, a, false, b,
                                                 (short)0, c, false, false);
}

// ---------------------------------------------------------------------------
// Kernel 1: per-key (column) softmax stats.  Grid: (b,h) x 8 k-chunks = 1024
// blocks of 256 threads (8 waves). Each wave owns 16 key columns and streams
// all 1024 queries through WMMA, maintaining online max / sum-exp.
// ---------------------------------------------------------------------------
__global__ __launch_bounds__(256) void attn_colstats_kernel(
    const float* __restrict__ x1, const float* __restrict__ x2,
    float* __restrict__ mG, float* __restrict__ lG) {
  __shared__ __attribute__((aligned(16))) unsigned short k1s[128 * LQ]; // [tok][d]
  __shared__ __attribute__((aligned(16))) unsigned short q2s[128 * LQ]; // [tok][d]

  const int tid  = threadIdx.x;
  const int lane = tid & 31;
  const int wave = tid >> 5;
  const int col  = lane & 15;      // N index (key column) / M index (query row)
  const int hiH  = lane >> 4;      // lane half selector

  const int bh = blockIdx.x >> 3;          // 0..127
  const int kc = blockIdx.x & 7;           // key chunk of 128
  const int b  = bh >> 3, h = bh & 7;
  const float* g1 = x1 + ((size_t)b * Cc + h * Dd) * Nn;  // [64][1024] chan-major
  const float* g2 = x2 + ((size_t)b * Cc + h * Dd) * Nn;
  const int kt0 = kc * 128;

  // Token-major staging: thread owns 32 d-values of one token; transpose on
  // the (coalesced) global-load side, packed ds_store_b64 on the LDS side.
  const int stok = tid & 127;
  const int sd0  = (tid >> 7) * 32;

#pragma unroll
  for (int j = 0; j < 32; j += 4) {
    const int dI = sd0 + j;
    const float* gp = g1 + (size_t)dI * Nn + kt0 + stok;
    v2u w; w.x = pk2bf(gp[0], gp[Nn]); w.y = pk2bf(gp[2 * Nn], gp[3 * Nn]);
    *(v2u*)(k1s + stok * LQ + dI) = w;
  }
  __syncthreads();

  // Resident B fragments of this wave's 16 key rows (K = d, 2 steps of 32).
  const unsigned short* krow = k1s + (wave * 16 + col) * LQ;
  v16bf bK[2];
#pragma unroll
  for (int s = 0; s < 2; ++s)
    bK[s] = ldsfrag(krow + s * 32 + hiH * 8, 0, 16);

  float m_run = -3.0e38f, l_run = 0.0f;

  for (int qc = 0; qc < 8; ++qc) {
    __syncthreads();
#pragma unroll
    for (int j = 0; j < 32; j += 4) {
      const int dI = sd0 + j;
      const float* gp = g2 + (size_t)dI * Nn + qc * 128 + stok;
      v2u w; w.x = pk2bf(gp[0], gp[Nn]); w.y = pk2bf(gp[2 * Nn], gp[3 * Nn]);
      *(v2u*)(q2s + stok * LQ + dI) = w;
    }
    __syncthreads();

#pragma unroll 2
    for (int qt = 0; qt < 8; ++qt) {
      const unsigned short* qrow = q2s + (qt * 16 + col) * LQ;
      v8f s = {};
#pragma unroll
      for (int ss = 0; ss < 2; ++ss) {
        v16bf a = ldsfrag(qrow + ss * 32 + hiH * 8, 0, 16);
        s = wmma_bf16(a, bK[ss], s);
      }
      // Column reduction: 8 rows in-lane, partner rows in lane^16.
      float sc[8];
#pragma unroll
      for (int r = 0; r < 8; ++r) sc[r] = s[r] * 0.125f;   // d^-0.5
      float tmax = sc[0];
#pragma unroll
      for (int r = 1; r < 8; ++r) tmax = fmaxf(tmax, sc[r]);
      tmax = fmaxf(tmax, __shfl_xor(tmax, 16, 32));
      float mnew = fmaxf(m_run, tmax);
      float psum = 0.0f;
#pragma unroll
      for (int r = 0; r < 8; ++r) psum += __expf(sc[r] - mnew);
      psum += __shfl_xor(psum, 16, 32);
      l_run = l_run * __expf(m_run - mnew) + psum;
      m_run = mnew;
    }
  }

  if (lane < 16) {
    int k = kt0 + wave * 16 + col;
    mG[bh * Nn + k] = m_run;
    lG[bh * Nn + k] = l_run;
  }
}

// ---------------------------------------------------------------------------
// Kernel 2: out = softmax_cols(S) @ V.  Grid: (b,h) x 8 q-chunks = 1024 blocks
// of 256 threads. Each wave owns 16 query rows and all 64 output dims; loops
// over keys in steps of 32 (one WMMA-K of the second GEMM). P tiles round-trip
// through a per-wave LDS buffer to convert C-layout -> A-layout.
// ---------------------------------------------------------------------------
__global__ __launch_bounds__(256) void attn_out_kernel(
    const float* __restrict__ x1, const float* __restrict__ x2,
    const float* __restrict__ mG, const float* __restrict__ lG,
    float* __restrict__ out) {
  __shared__ __attribute__((aligned(16))) unsigned short q2s[128 * LQ];     // [q][d]
  __shared__ __attribute__((aligned(16))) unsigned short k1t[32 * LQ];      // [tok][d]
  __shared__ __attribute__((aligned(16))) unsigned short v2t[Dd * LV];      // [d][tok]
  __shared__ __attribute__((aligned(16))) unsigned short ps[8 * 16 * LV];   // per-wave P [q][k]

  const int tid  = threadIdx.x;
  const int lane = tid & 31;
  const int wave = tid >> 5;
  const int col  = lane & 15;
  const int hiH  = lane >> 4;

  const int bh = blockIdx.x >> 3;
  const int qc = blockIdx.x & 7;
  const int b  = bh >> 3, h = bh & 7;
  const float* g1 = x1 + ((size_t)b * Cc + h * Dd) * Nn;
  const float* g2 = x2 + ((size_t)b * Cc + h * Dd) * Nn;
  const int q0 = qc * 128;

  // Stage this block's 128 query rows (token-major, packed b64 stores).
  {
    const int stok = tid & 127;
    const int sd0  = (tid >> 7) * 32;
#pragma unroll
    for (int j = 0; j < 32; j += 4) {
      const int dI = sd0 + j;
      const float* gp = g2 + (size_t)dI * Nn + q0 + stok;
      v2u w; w.x = pk2bf(gp[0], gp[Nn]); w.y = pk2bf(gp[2 * Nn], gp[3 * Nn]);
      *(v2u*)(q2s + stok * LQ + dI) = w;
    }
  }
  __syncthreads();

  // Resident A fragments of this wave's 16 query rows (K = d).
  const unsigned short* qrow = q2s + (wave * 16 + col) * LQ;
  v16bf aQ[2];
#pragma unroll
  for (int s = 0; s < 2; ++s)
    aQ[s] = ldsfrag(qrow + s * 32 + hiH * 8, 0, 16);

  v8f acc[4] = {};
  const float* mRow = mG + bh * Nn;
  const float* lRow = lG + bh * Nn;
  unsigned short* pw = ps + wave * 16 * LV;

  // Staging thread mappings for the per-step tiles.
  const int ttok = tid & 31;             // k1t: token-major transpose copy
  const int td0  = (tid >> 5) * 8;
  const int vtok = (tid & 7) * 4;        // v2t: layout matches global [d][tok]
  const int vd0  = tid >> 3;

  for (int ks = 0; ks < 32; ++ks) {
    const int kt = ks * 32;
    __syncthreads();
    // k1 tile -> [tok][d] (transposed); v2 tile -> [d][tok] (direct, b128).
#pragma unroll
    for (int j = 0; j < 8; j += 4) {
      const int dI = td0 + j;
      const float* gp = g1 + (size_t)dI * Nn + kt + ttok;
      v2u w; w.x = pk2bf(gp[0], gp[Nn]); w.y = pk2bf(gp[2 * Nn], gp[3 * Nn]);
      *(v2u*)(k1t + ttok * LQ + dI) = w;
    }
#pragma unroll
    for (int rr = 0; rr < 2; ++rr) {
      const int dI = vd0 + rr * 32;
      v4f f = *(const v4f*)(g2 + (size_t)dI * Nn + kt + vtok);
      v2u w; w.x = pk2bf(f.x, f.y); w.y = pk2bf(f.z, f.w);
      *(v2u*)(v2t + dI * LV + vtok) = w;
    }
    __syncthreads();

    // GEMM 1: S tiles (16q x 16k) x2, then normalize to P and stage to LDS.
#pragma unroll
    for (int t = 0; t < 2; ++t) {
      const unsigned short* krow = k1t + (t * 16 + col) * LQ;
      v8f s = {};
#pragma unroll
      for (int ss = 0; ss < 2; ++ss) {
        v16bf bk = ldsfrag(krow + ss * 32 + hiH * 8, 0, 16);
        s = wmma_bf16(aQ[ss], bk, s);
      }
      const int kcol = kt + t * 16 + col;
      const float mv = mRow[kcol];
      const float rl = 1.0f / lRow[kcol];
#pragma unroll
      for (int r = 0; r < 8; ++r) {
        float p = __expf(s[r] * 0.125f - mv) * rl;
        pw[(r + hiH * 8) * LV + t * 16 + col] = f2bf(p);  // C-layout -> [q][k]
      }
    }
    // Per-wave staging buffer: order LDS stores before fragment reloads.
    asm volatile("s_wait_dscnt 0" ::: "memory");

    // GEMM 2: out += P(16x32) @ V(32x64)
    v16bf aP = ldsfrag(pw + col * LV + hiH * 8, 0, 16);
#pragma unroll
    for (int t = 0; t < 4; ++t) {
      v16bf bv = ldsfrag(v2t + (t * 16 + col) * LV + hiH * 8, 0, 16);
      acc[t] = wmma_bf16(aP, bv, acc[t]);
    }
  }

  // Epilogue: out[b, q0 + wave*16 + M, h*64 + t*16 + col], f32.
  float* orow = out + ((size_t)(b * Nn + q0 + wave * 16)) * Cc + h * Dd;
#pragma unroll
  for (int t = 0; t < 4; ++t)
#pragma unroll
    for (int r = 0; r < 8; ++r)
      orow[(size_t)(r + hiH * 8) * Cc + t * 16 + col] = acc[t][r];
}

// ---------------------------------------------------------------------------
extern "C" void kernel_launch(void* const* d_in, const int* in_sizes, int n_in,
                              void* d_out, int out_size, void* d_ws, size_t ws_size,
                              hipStream_t stream) {
  (void)in_sizes; (void)n_in; (void)out_size; (void)ws_size;
  const float* x1 = (const float*)d_in[0];
  const float* x2 = (const float*)d_in[1];
  // d_in[2] = num_heads (always 8 per setup_inputs)
  float* mG = (float*)d_ws;                      // [128][1024] column maxes
  float* lG = mG + (size_t)Bn * Hn * Nn;         // [128][1024] column sum-exp
  float* outp = (float*)d_out;

  dim3 grid(Bn * Hn * (Nn / 128));               // 1024 blocks
  dim3 block(256);                               // 8 waves (wave32)
  attn_colstats_kernel<<<grid, block, 0, stream>>>(x1, x2, mG, lG);
  attn_out_kernel<<<grid, block, 0, stream>>>(x1, x2, mG, lG, outp);
}
